// HetAgg_18554258719139
// MI455X (gfx1250) — matrix operand order repaired and checked by reading, not compiled, not run on hardware
//
#include <hip/hip_runtime.h>
#include <hip/hip_bf16.h>

typedef __attribute__((ext_vector_type(16))) _Float16 v16h;
typedef __attribute__((ext_vector_type(8)))  _Float16 v8h;
typedef __attribute__((ext_vector_type(8)))  float    v8f;

#define T_  11
#define NN_ 50000
#define D_  128
#define H_  64
#define B_  16384
#define K_  10

static __device__ __forceinline__ float sigm(float x) { return 1.0f / (1.0f + __expf(-x)); }

static __device__ __forceinline__ v8f wmma16(v16h a, v16h b, v8f c) {
    return __builtin_amdgcn_wmma_f32_16x16x32_f16(false, a, false, b, (short)0, c, false, false);
}

// A-fragment (16x32 f16) from row-major f32 row:
// lanes 0-15 hold K = k0+{0..15}; lanes 16-31 hold K = k0+{16..31}.
static __device__ __forceinline__ v16h packA_f32(const float* __restrict__ row, int k0, int hi) {
    const float4* q0 = (const float4*)(row + k0 + 16 * hi);
    const float4* q1 = (const float4*)(row + k0 + 16 * hi + 8);
    float4 x0 = q0[0], x1 = q0[1], y0 = q1[0], y1 = q1[1];
    v16h a;
    a[0] = (_Float16)x0.x; a[1] = (_Float16)x0.y; a[2] = (_Float16)x0.z; a[3] = (_Float16)x0.w;
    a[4] = (_Float16)x1.x; a[5] = (_Float16)x1.y; a[6] = (_Float16)x1.z; a[7] = (_Float16)x1.w;
    a[8] = (_Float16)y0.x; a[9] = (_Float16)y0.y; a[10] = (_Float16)y0.z; a[11] = (_Float16)y0.w;
    a[12] = (_Float16)y1.x; a[13] = (_Float16)y1.y; a[14] = (_Float16)y1.z; a[15] = (_Float16)y1.w;
    return a;
}

static __device__ __forceinline__ v16h packA_f16(const _Float16* __restrict__ row, int k0, int hi) {
    v8h a0 = *(const v8h*)(row + k0 + 16 * hi);
    v8h a1 = *(const v8h*)(row + k0 + 16 * hi + 8);
    v16h a;
#pragma unroll
    for (int i = 0; i < 8; ++i) { a[i] = a0[i]; a[8 + i] = a1[i]; }
    return a;
}

// ---------------------------------------------------------------------------
// Pack fwd/bwd weight pair into one contiguous buffer:
// out layout [T][512][rowLen]; rows 0-255 from fwd, 256-511 from bwd.
// ---------------------------------------------------------------------------
__global__ void pack2_f16_kernel(const float* __restrict__ fwd, const float* __restrict__ bwd,
                                 _Float16* __restrict__ out, int rowLen, int n) {
    int i = blockIdx.x * 256 + threadIdx.x;
    if (i >= n) return;
    int k = i % rowLen;
    int r = (i / rowLen) & 511;
    int t = i / (rowLen * 512);
    float v = (r < 256) ? fwd[((size_t)t * 256 + r) * rowLen + k]
                        : bwd[((size_t)t * 256 + (r - 256)) * rowLen + k];
    out[i] = (_Float16)v;
}

__global__ void pack2_f32_kernel(const float* __restrict__ fwd, const float* __restrict__ bwd,
                                 float* __restrict__ out, int rowLen, int n) {
    int i = blockIdx.x * 256 + threadIdx.x;
    if (i >= n) return;
    int k = i % rowLen;
    int r = (i / rowLen) & 511;
    int t = i / (rowLen * 512);
    out[i] = (r < 256) ? fwd[((size_t)t * 256 + r) * rowLen + k]
                       : bwd[((size_t)t * 256 + (r - 256)) * rowLen + k];
}

// ---------------------------------------------------------------------------
// Zero-state bi-LSTM cell on gathered rows (h=0 => gates = x@Wih.T + b).
// Workgroup: 16 rows x 512 gates (fwd 256 | bwd 256), 8 waves x 4 WMMA tiles.
// wC: packed [512][128] f16 for this type; bC: packed [512] f32.
// ---------------------------------------------------------------------------
__global__ void content_encode_kernel(const float* __restrict__ feat,
                                      const int* __restrict__ nids,
                                      const _Float16* __restrict__ wC,
                                      const float* __restrict__ bC,
                                      _Float16* __restrict__ outH,
                                      float* __restrict__ outF,
                                      int selfMode) {
    __shared__ float G[16][512];

    const int m0 = blockIdx.x * 16;
    const int lane = threadIdx.x & 31;
    const int wv = threadIdx.x >> 5;
    const int hi = lane >> 4;
    const int ln = lane & 15;

    const int row = m0 + ln;
    int nid;
    if (selfMode) {
        nid = nids[row];
    } else {
        const int kk = row >> 14;      // / B_
        const int b = row & (B_ - 1);
        nid = nids[b * K_ + kk];
    }
    const float* arow = feat + (size_t)nid * D_;

    v8f acc[4] = {{}, {}, {}, {}};
    const int gb = wv * 64;  // this wave's 4 N-tiles

#pragma unroll
    for (int c = 0; c < 4; ++c) {
        const int k0 = c * 32;
        v16h a = packA_f32(arow, k0, hi);
#pragma unroll
        for (int j = 0; j < 4; ++j) {
            const _Float16* W = wC + (size_t)(gb + j * 16 + ln) * D_ + k0 + 16 * hi;
            v16h bf = *(const v16h*)W;
            acc[j] = wmma16(a, bf, acc[j]);
        }
    }
#pragma unroll
    for (int j = 0; j < 4; ++j)
#pragma unroll
        for (int r = 0; r < 8; ++r)
            G[r + 8 * hi][gb + j * 16 + ln] = acc[j][r];
    __syncthreads();

    // epilogue: 16 rows x 128 outputs (fwd 64 | bwd 64) = 2048 -> 8 per thread
#pragma unroll
    for (int it = 0; it < 8; ++it) {
        const int idx = threadIdx.x + it * 256;
        const int mm = idx >> 7;
        const int jj = idx & 127;
        const int dir = jj >> 6;
        const int j = jj & 63;
        const int base = dir * 256;
        const float gi = G[mm][base + j] + bC[base + j];
        const float gg = G[mm][base + 128 + j] + bC[base + 128 + j];
        const float go = G[mm][base + 192 + j] + bC[base + 192 + j];
        const float cc = sigm(gi) * tanhf(gg);
        const float h = sigm(go) * tanhf(cc);
        const size_t grow = (size_t)(m0 + mm);
        if (selfMode) outF[grow * D_ + jj] = h;
        else outH[grow * D_ + jj] = (_Float16)h;
    }
}

// ---------------------------------------------------------------------------
// One sequential bi-LSTM step; fwd/bwd via blockIdx.y (scalar offset into the
// packed weights). Workgroup: 16 rows x 256 gates, 8 waves x 2 WMMA tiles.
// ---------------------------------------------------------------------------
__global__ void lstm_step_kernel(const _Float16* __restrict__ ce,     // [K][B][128] f16
                                 const _Float16* __restrict__ wNiT,   // packed [512][128] f16
                                 const _Float16* __restrict__ wNhT,   // packed [512][64] f16
                                 const float* __restrict__ bNT,       // packed [512] f32
                                 const _Float16* __restrict__ hPrev,  // [2][B][64] f16
                                 _Float16* __restrict__ hOut,         // [2][B][64] f16
                                 float* __restrict__ cState,          // [2][B][64] f32
                                 float* __restrict__ aggT,            // [B][128] f32
                                 int step) {
    __shared__ float G[16][256];

    const int dir = blockIdx.y;
    const int b0 = blockIdx.x * 16;
    const int xk = dir ? (K_ - 1 - step) : step;
    const _Float16* x = ce + (size_t)xk * B_ * D_;
    const _Float16* wIh = wNiT + (size_t)dir * 256 * D_;
    const _Float16* wHh = wNhT + (size_t)dir * 256 * H_;
    const float* bias = bNT + (size_t)dir * 256;
    const _Float16* hp = hPrev + (size_t)dir * B_ * H_;

    const int lane = threadIdx.x & 31;
    const int wv = threadIdx.x >> 5;
    const int hi = lane >> 4;
    const int ln = lane & 15;
    const int row = b0 + ln;
    const _Float16* xrow = x + (size_t)row * D_;

    v8f acc0 = {}, acc1 = {};
    const int gb = wv * 32;  // two tiles: gb, gb+16

    // x contribution: K = 128
#pragma unroll
    for (int c = 0; c < 4; ++c) {
        const int k0 = c * 32;
        v16h a = packA_f16(xrow, k0, hi);
        v16h bf0 = *(const v16h*)(wIh + (size_t)(gb + ln) * D_ + k0 + 16 * hi);
        v16h bf1 = *(const v16h*)(wIh + (size_t)(gb + 16 + ln) * D_ + k0 + 16 * hi);
        acc0 = wmma16(a, bf0, acc0);
        acc1 = wmma16(a, bf1, acc1);
    }
    // h contribution: K = 64 (skip at step 0: h == 0); uniform branch
    if (step > 0) {
        const _Float16* hrow = hp + (size_t)row * H_;
#pragma unroll
        for (int c = 0; c < 2; ++c) {
            const int k0 = c * 32;
            v16h a = packA_f16(hrow, k0, hi);
            v16h bf0 = *(const v16h*)(wHh + (size_t)(gb + ln) * H_ + k0 + 16 * hi);
            v16h bf1 = *(const v16h*)(wHh + (size_t)(gb + 16 + ln) * H_ + k0 + 16 * hi);
            acc0 = wmma16(a, bf0, acc0);
            acc1 = wmma16(a, bf1, acc1);
        }
    }
#pragma unroll
    for (int r = 0; r < 8; ++r) {
        G[r + 8 * hi][gb + ln] = acc0[r];
        G[r + 8 * hi][gb + 16 + ln] = acc1[r];
    }
    __syncthreads();

    // epilogue: 16 rows x 64 hidden -> 4 per thread
#pragma unroll
    for (int it = 0; it < 4; ++it) {
        const int idx = threadIdx.x + it * 256;
        const int mm = idx >> 6;
        const int j = idx & 63;
        const float gi = G[mm][j] + bias[j];
        const float gf = G[mm][64 + j] + bias[64 + j];
        const float gg = G[mm][128 + j] + bias[128 + j];
        const float go = G[mm][192 + j] + bias[192 + j];
        const size_t srow = (size_t)(b0 + mm);
        float* cp = cState + (size_t)dir * B_ * H_ + srow * H_ + j;
        const float cprev = (step > 0) ? *cp : 0.0f;
        const float cn = sigm(gf) * cprev + sigm(gi) * tanhf(gg);
        const float h = sigm(go) * tanhf(cn);
        *cp = cn;
        hOut[(size_t)dir * B_ * H_ + srow * H_ + j] = (_Float16)h;
        float* ag = aggT + srow * D_ + dir * H_ + j;
        if (step == 0) *ag = h;
        else *ag += h;
    }
}

// ---------------------------------------------------------------------------
// Attention combine: 12 embeds (c_self + 11 type aggs / K), leaky-relu scores,
// softmax, weighted sum. One 128-thread block per batch row.
// ---------------------------------------------------------------------------
__global__ void attention_kernel(const float* __restrict__ cself,  // [B][128]
                                 const float* __restrict__ agg,    // [T][B][128]
                                 const float* __restrict__ att,    // [256]
                                 float* __restrict__ out) {        // [B][128]
    __shared__ float red[13][128];
    __shared__ float s[13];
    __shared__ float w[12];

    const int b = blockIdx.x;
    const int d = threadIdx.x;
    const float a0 = cself[(size_t)b * D_ + d];
    const float alo = att[d];
    const float ahi = att[D_ + d];

    float e[12];
    e[0] = a0;
#pragma unroll
    for (int t = 1; t < 12; ++t)
        e[t] = agg[(size_t)(t - 1) * B_ * D_ + (size_t)b * D_ + d] * (1.0f / (float)K_);

    red[0][d] = a0 * alo;
#pragma unroll
    for (int t = 0; t < 12; ++t) red[1 + t][d] = e[t] * ahi;
    __syncthreads();

    if (d < 13) {
        float sum = 0.0f;
        for (int i = 0; i < 128; ++i) sum += red[d][i];
        s[d] = sum;
    }
    __syncthreads();

    if (d == 0) {
        float sc[12];
        float mx = -3.4e38f;
        for (int t = 0; t < 12; ++t) {
            float v = s[0] + s[1 + t];
            v = (v > 0.0f) ? v : 0.01f * v;  // leaky relu
            sc[t] = v;
            mx = fmaxf(mx, v);
        }
        float ss = 0.0f;
        for (int t = 0; t < 12; ++t) { sc[t] = __expf(sc[t] - mx); ss += sc[t]; }
        const float inv = 1.0f / ss;
        for (int t = 0; t < 12; ++t) w[t] = sc[t] * inv;
    }
    __syncthreads();

    float o = 0.0f;
#pragma unroll
    for (int t = 0; t < 12; ++t) o += w[t] * e[t];
    out[(size_t)b * D_ + d] = o;
}

// ---------------------------------------------------------------------------
// host
// ---------------------------------------------------------------------------
extern "C" void kernel_launch(void* const* d_in, const int* in_sizes, int n_in,
                              void* d_out, int out_size, void* d_ws, size_t ws_size,
                              hipStream_t stream) {
    const float* features = (const float*)d_in[0];
    const int* id_batch = (const int*)d_in[1];
    const int* neigh = (const int*)d_in[2];
    const float* cWih_f = (const float*)d_in[3];
    // d_in[4] = cWhh_f (dead: h==0 in bi_step)
    const float* cb_f = (const float*)d_in[5];
    const float* cWih_b = (const float*)d_in[6];
    // d_in[7] = cWhh_b (dead)
    const float* cb_b = (const float*)d_in[8];
    const float* nWih_f = (const float*)d_in[9];
    const float* nWhh_f = (const float*)d_in[10];
    const float* nb_f = (const float*)d_in[11];
    const float* nWih_b = (const float*)d_in[12];
    const float* nWhh_b = (const float*)d_in[13];
    const float* nb_b = (const float*)d_in[14];
    const float* att = (const float*)d_in[15];

    const int nIhP = T_ * 512 * D_;  // packed Wih elements: 720896
    const int nHhP = T_ * 512 * H_;  // packed Whh elements: 360448
    const int nBP = T_ * 512;        // packed bias elements: 5632

    char* ws = (char*)d_ws;
    size_t off = 0;
    auto carve = [&](size_t bytes) { size_t o = off; off += (bytes + 255) & ~(size_t)255; return o; };
    _Float16* wC = (_Float16*)(ws + carve((size_t)nIhP * 2));   // content Wih packed [T][512][128]
    _Float16* wNi = (_Float16*)(ws + carve((size_t)nIhP * 2));  // neigh Wih packed
    _Float16* wNh = (_Float16*)(ws + carve((size_t)nHhP * 2));  // neigh Whh packed [T][512][64]
    float* bC = (float*)(ws + carve((size_t)nBP * 4));          // content bias packed [T][512]
    float* bN = (float*)(ws + carve((size_t)nBP * 4));          // neigh bias packed
    float* cself = (float*)(ws + carve((size_t)B_ * D_ * 4));
    _Float16* ceBuf = (_Float16*)(ws + carve((size_t)K_ * B_ * D_ * 2));  // reused per type
    _Float16* hPP = (_Float16*)(ws + carve((size_t)2 * 2 * B_ * H_ * 2)); // ping-pong [2][2dir][B][64]
    float* cSt = (float*)(ws + carve((size_t)2 * B_ * H_ * 4));           // [2dir][B][64]
    float* agg = (float*)(ws + carve((size_t)T_ * B_ * D_ * 4));          // [T][B][128]

    // 1) pack + convert live weights (cWhh is dead: h==0 in bi_step)
    pack2_f16_kernel<<<(nIhP + 255) / 256, 256, 0, stream>>>(cWih_f, cWih_b, wC, D_, nIhP);
    pack2_f16_kernel<<<(nIhP + 255) / 256, 256, 0, stream>>>(nWih_f, nWih_b, wNi, D_, nIhP);
    pack2_f16_kernel<<<(nHhP + 255) / 256, 256, 0, stream>>>(nWhh_f, nWhh_b, wNh, H_, nHhP);
    pack2_f32_kernel<<<(nBP + 255) / 256, 256, 0, stream>>>(cb_f, cb_b, bC, 1, nBP);
    pack2_f32_kernel<<<(nBP + 255) / 256, 256, 0, stream>>>(nb_f, nb_b, bN, 1, nBP);

    // 2) self content embedding (type-0 weights on id_batch rows)
    content_encode_kernel<<<B_ / 16, 256, 0, stream>>>(
        features, id_batch, wC, bC, (_Float16*)nullptr, cself, 1);

    // 3) per-type: content-encode neighbors, then K sequential bi-LSTM steps
    for (int t = 0; t < T_; ++t) {
        content_encode_kernel<<<(K_ * B_) / 16, 256, 0, stream>>>(
            features + (size_t)t * NN_ * D_, neigh + (size_t)t * B_ * K_,
            wC + (size_t)t * 512 * D_, bC + (size_t)t * 512,
            ceBuf, (float*)nullptr, 0);

        for (int k = 0; k < K_; ++k) {
            _Float16* hOut = hPP + (size_t)(k & 1) * 2 * B_ * H_;
            _Float16* hIn = hPP + (size_t)((k + 1) & 1) * 2 * B_ * H_;
            lstm_step_kernel<<<dim3(B_ / 16, 2), 256, 0, stream>>>(
                ceBuf,
                wNi + (size_t)t * 512 * D_, wNh + (size_t)t * 512 * H_,
                bN + (size_t)t * 512,
                hIn, hOut, cSt, agg + (size_t)t * B_ * D_, k);
        }
    }

    // 4) attention combine
    attention_kernel<<<B_, 128, 0, stream>>>(cself, agg, att, (float*)d_out);

    (void)in_sizes; (void)n_in; (void)out_size; (void)ws_size;
}
